// DOFEN_52020643889652
// MI455X (gfx1250) — matrix operand order repaired and checked by reading, not compiled, not run on hardware
//
#include <hip/hip_runtime.h>
#include <hip/hip_bf16.h>

#define B_      1024
#define NCOL    64
#define NCOND   64
#define NRODT   1024
#define D_      4
#define NEST    128
#define NFOREST 100
#define NHID    128
#define NCLASS  10
#define EPS_    1e-5f

typedef _Float16 v16h __attribute__((ext_vector_type(16)));
typedef _Float16 v8h  __attribute__((ext_vector_type(8)));
typedef float    v8f  __attribute__((ext_vector_type(8)));

// ---------------------------------------------------------------------------
// Prep 1: fold fixed permutation into gathered w1/b1/col tables.
// O[b,j] = sigmoid(x[b,c]*w1[c,0,k] + b1[c*64+k]) with p=perm[j], k=p>>6, c=p&63.
// ---------------------------------------------------------------------------
__global__ void k_prep_perm(const int* __restrict__ perm,
                            const float* __restrict__ w1,
                            const float* __restrict__ b1,
                            float* __restrict__ wp, float* __restrict__ bp,
                            int* __restrict__ cix) {
    int j = blockIdx.x * blockDim.x + threadIdx.x;
    if (j >= NCOL * NCOND) return;
    int p  = perm[j];
    int c  = p & (NCOL - 1);
    int k  = p >> 6;
    int wi = c * NCOND + k;
    wp[j]  = w1[wi];
    bp[j]  = b1[wi];
    cix[j] = c;
}

// ---------------------------------------------------------------------------
// Prep 2: EsT16[f][n][k] = (f16) E[swr[f,k], n]   (B-operand transposed layout)
// ---------------------------------------------------------------------------
__global__ void k_prep_est(const float* __restrict__ E, const int* __restrict__ swr,
                           _Float16* __restrict__ EsT) {
    int idx = blockIdx.x * blockDim.x + threadIdx.x;   // f*16384 + n*128 + k
    if (idx >= NFOREST * NHID * NEST) return;
    int k = idx & 127;
    int n = (idx >> 7) & 127;
    int f = idx >> 14;
    int row = swr[f * NEST + k];
    EsT[idx] = (_Float16)E[row * NHID + n];
}

// ---------------------------------------------------------------------------
// Prep 3: W1T16[n][k] = (f16) lin1_w[k,n]
// ---------------------------------------------------------------------------
__global__ void k_prep_w1t(const float* __restrict__ lin1_w, _Float16* __restrict__ W1T) {
    int idx = blockIdx.x * blockDim.x + threadIdx.x;   // n*128 + k
    if (idx >= NHID * NHID) return;
    int k = idx & 127;
    int n = idx >> 7;
    W1T[idx] = (_Float16)lin1_w[k * NHID + n];
}

// ---------------------------------------------------------------------------
// Fused phi_1 + rODT + phi_2: one thread per (batch b, group g) -> w[b,g].
// Per-group work is D=4 wide: pure VALU (no WMMA shape fits K=4 usefully).
// ---------------------------------------------------------------------------
__global__ void k_rodt(const float* __restrict__ x,
                       const float* __restrict__ wp, const float* __restrict__ bp,
                       const int* __restrict__ cix,
                       const float* __restrict__ gn1_g, const float* __restrict__ gn1_b,
                       const float* __restrict__ w2a, const float* __restrict__ b2a,
                       const float* __restrict__ gn2_g, const float* __restrict__ gn2_b,
                       const float* __restrict__ w2b, const float* __restrict__ b2b,
                       float* __restrict__ wbuf) {
    int idx = blockIdx.x * blockDim.x + threadIdx.x;
    if (idx >= B_ * NRODT) return;
    int b  = idx >> 10;
    int g  = idx & (NRODT - 1);
    int j0 = g * D_;

    float v[D_];
    float s = 0.f;
#pragma unroll
    for (int i = 0; i < D_; i++) {
        int j   = j0 + i;
        float t = x[b * NCOL + cix[j]] * wp[j] + bp[j];
        t = 1.0f / (1.0f + __expf(-t));      // sigmoid
        v[i] = t; s += t;
    }
    float mu = s * 0.25f, var = 0.f;
#pragma unroll
    for (int i = 0; i < D_; i++) { float d = v[i] - mu; var += d * d; }
    var *= 0.25f;
    float rstd = rsqrtf(var + EPS_);

    float vn[D_];
#pragma unroll
    for (int i = 0; i < D_; i++)
        vn[i] = (v[i] - mu) * rstd * gn1_g[j0 + i] + gn1_b[j0 + i];

    float h[D_]; float s2 = 0.f;
#pragma unroll
    for (int o = 0; o < D_; o++) {
        float acc = b2a[j0 + o];
#pragma unroll
        for (int i = 0; i < D_; i++)
            acc += vn[i] * w2a[g * (D_ * D_) + i * D_ + o];
        acc = fmaxf(acc, 0.f);
        h[o] = acc; s2 += acc;
    }
    float mu2 = s2 * 0.25f, var2 = 0.f;
#pragma unroll
    for (int o = 0; o < D_; o++) { float d = h[o] - mu2; var2 += d * d; }
    var2 *= 0.25f;
    float rstd2 = rsqrtf(var2 + EPS_);

    float wv = b2b[g];
#pragma unroll
    for (int o = 0; o < D_; o++) {
        float hn = (h[o] - mu2) * rstd2 * gn2_g[j0 + o] + gn2_b[j0 + o];
        wv += hn * w2b[j0 + o];
    }
    wbuf[idx] = wv;
}

__global__ void k_zero(float* __restrict__ p, int n) {
    int i = blockIdx.x * blockDim.x + threadIdx.x;
    if (i < n) p[i] = 0.f;
}

// ---------------------------------------------------------------------------
// Fused forest + phi_3 MLP: block = (forest f, 32-row batch tile), 256 threads
// (8 waves). Both weight tiles (Es^T and lin1_w^T) are staged into separate
// LDS buffers up-front so the global->LDS latency overlaps the softmax VALU
// phase and GEMM2 never stalls on a mid-kernel reload.
// softmax -> WMMA GEMM (ws @ Es) -> LN -> WMMA GEMM (lin1) -> ReLU -> LN
// -> 128x10 head -> atomic accumulate mean over forests.
// ---------------------------------------------------------------------------
__global__ __launch_bounds__(256, 2)
void k_forest(const float* __restrict__ wbuf, const int* __restrict__ swr,
              const _Float16* __restrict__ EsT, const _Float16* __restrict__ W1T,
              const float* __restrict__ ln1_g, const float* __restrict__ ln1_b,
              const float* __restrict__ lin1_b,
              const float* __restrict__ ln2_g, const float* __restrict__ ln2_b,
              const float* __restrict__ lin2_w, const float* __restrict__ lin2_b,
              float* __restrict__ out) {
    __shared__ __attribute__((aligned(32))) _Float16 sWS[32][128];  // A operand (f16)
    __shared__ __attribute__((aligned(32))) _Float16 sBE[128][128]; // Es^T  [n][k] (f16)
    __shared__ __attribute__((aligned(32))) _Float16 sBW[128][128]; // lin1^T[n][k] (f16)
    __shared__ float sF[32][128];                                   // f32 activations
    __shared__ float sred[2][32][8];
    __shared__ float smean[32], srstd[32];
    __shared__ int   ssw[128];
    __shared__ float sg1[128], sb1g[128], sbias[128], sg2[128], sb2g[128];

    const int tid = threadIdx.x;
    const int f   = blockIdx.x;
    const int b0  = blockIdx.y * 32;

    // warm L2 for broadcast-read weight tables (global_prefetch_b8)
    __builtin_prefetch(W1T + tid * 64, 0, 1);
    __builtin_prefetch(lin2_w + (tid & 63) * 20, 0, 1);

    if (tid < 128) {
        ssw[tid]  = swr[f * NEST + tid];
        sg1[tid]  = ln1_g[tid];  sb1g[tid] = ln1_b[tid];
        sbias[tid]= lin1_b[tid];
        sg2[tid]  = ln2_g[tid];  sb2g[tid] = ln2_b[tid];
    }
    __syncthreads();   // ssw ready for the softmax gather

    // ---- stage both B operands into LDS (overlaps with softmax below) ----
    {
        const unsigned int* ge = (const unsigned int*)(EsT + (size_t)f * NHID * NEST);
        unsigned int*       se = (unsigned int*)&sBE[0][0];
        for (int i = tid; i < NHID * NEST / 2; i += 256) se[i] = ge[i];
        const unsigned int* gw = (const unsigned int*)W1T;
        unsigned int*       sw = (unsigned int*)&sBW[0][0];
        for (int i = tid; i < NHID * NHID / 2; i += 256) sw[i] = gw[i];
    }

    const int r = tid >> 3, oct = tid & 7;           // 8 threads per batch row

    // ---- gather w[:, swr[f]] and softmax over 128 estimators ----
    {
        float mx = -1e30f;
#pragma unroll
        for (int jj = 0; jj < 16; jj++) {
            int e = oct * 16 + jj;
            float v = wbuf[(b0 + r) * NRODT + ssw[e]];
            sF[r][e] = v;
            mx = fmaxf(mx, v);
        }
        sred[0][r][oct] = mx;
        __syncthreads();
        float rm = sred[0][r][0];
#pragma unroll
        for (int q = 1; q < 8; q++) rm = fmaxf(rm, sred[0][r][q]);
        float s = 0.f;
#pragma unroll
        for (int jj = 0; jj < 16; jj++) {
            int e = oct * 16 + jj;
            float v = __expf(sF[r][e] - rm);
            sF[r][e] = v; s += v;
        }
        __syncthreads();
        sred[0][r][oct] = s;
        __syncthreads();
        float tot = 0.f;
#pragma unroll
        for (int q = 0; q < 8; q++) tot += sred[0][r][q];
        float inv = 1.0f / tot;
#pragma unroll
        for (int jj = 0; jj < 16; jj++) {
            int e = oct * 16 + jj;
            sWS[r][e] = (_Float16)(sF[r][e] * inv);
        }
    }
    __syncthreads();   // sWS + sBE + sBW all resident

    const int lane = tid & 31, wave = tid >> 5;
    const int ln15 = lane & 15, hi = lane >> 4;

    // ---- GEMM1: F(32x128) = ws(32x128) @ Es(128x128), K in 4 chunks of 32 ----
#pragma unroll
    for (int tt = 0; tt < 2; tt++) {
        int t  = wave * 2 + tt;                 // 16 output tiles (2 x 8)
        int mb = (t >> 3) * 16, nb = (t & 7) * 16;
        v8f acc = {0.f, 0.f, 0.f, 0.f, 0.f, 0.f, 0.f, 0.f};
#pragma unroll
        for (int kc = 0; kc < 4; kc++) {
            const _Float16* pA = &sWS[mb + ln15][kc * 32 + hi * 16];
            v8h alo = *(const v8h*)pA;
            v8h ahi = *(const v8h*)(pA + 8);
            v16h a  = __builtin_shufflevector(alo, ahi,
                         0,1,2,3,4,5,6,7,8,9,10,11,12,13,14,15);
            const _Float16* pB = &sBE[nb + ln15][kc * 32 + hi * 16];
            v16h bf = *(const v16h*)pB;
            acc = __builtin_amdgcn_wmma_f32_16x16x32_f16(
                      false, a, false, bf, (short)0, acc, false, false);
        }
#pragma unroll
        for (int v = 0; v < 8; v++)
            sF[mb + hi * 8 + v][nb + ln15] = acc[v];
    }
    __syncthreads();

    // ---- LayerNorm(F) * ln1 -> f16 into sWS ----
    {
        float s1 = 0.f, s2 = 0.f;
#pragma unroll
        for (int jj = 0; jj < 16; jj++) {
            int e = oct * 16 + jj;
            float v = sF[r][e]; s1 += v; s2 += v * v;
        }
        sred[0][r][oct] = s1; sred[1][r][oct] = s2;
        __syncthreads();
        float t1 = 0.f, t2 = 0.f;
#pragma unroll
        for (int q = 0; q < 8; q++) { t1 += sred[0][r][q]; t2 += sred[1][r][q]; }
        float mu = t1 * (1.0f / NHID);
        float rs = rsqrtf(t2 * (1.0f / NHID) - mu * mu + EPS_);
#pragma unroll
        for (int jj = 0; jj < 16; jj++) {
            int e = oct * 16 + jj;
            sWS[r][e] = (_Float16)((sF[r][e] - mu) * rs * sg1[e] + sb1g[e]);
        }
    }
    __syncthreads();

    // ---- GEMM2: z @ lin1_w, + bias, ReLU -> sF (sBW already resident) ----
#pragma unroll
    for (int tt = 0; tt < 2; tt++) {
        int t  = wave * 2 + tt;
        int mb = (t >> 3) * 16, nb = (t & 7) * 16;
        v8f acc = {0.f, 0.f, 0.f, 0.f, 0.f, 0.f, 0.f, 0.f};
#pragma unroll
        for (int kc = 0; kc < 4; kc++) {
            const _Float16* pA = &sWS[mb + ln15][kc * 32 + hi * 16];
            v8h alo = *(const v8h*)pA;
            v8h ahi = *(const v8h*)(pA + 8);
            v16h a  = __builtin_shufflevector(alo, ahi,
                         0,1,2,3,4,5,6,7,8,9,10,11,12,13,14,15);
            const _Float16* pB = &sBW[nb + ln15][kc * 32 + hi * 16];
            v16h bf = *(const v16h*)pB;
            acc = __builtin_amdgcn_wmma_f32_16x16x32_f16(
                      false, a, false, bf, (short)0, acc, false, false);
        }
#pragma unroll
        for (int v = 0; v < 8; v++) {
            int mm = mb + hi * 8 + v, nn = nb + ln15;
            sF[mm][nn] = fmaxf(acc[v] + sbias[nn], 0.f);
        }
    }
    __syncthreads();

    // ---- LayerNorm2 stats per row ----
    {
        float s1 = 0.f, s2 = 0.f;
#pragma unroll
        for (int jj = 0; jj < 16; jj++) {
            int e = oct * 16 + jj;
            float v = sF[r][e]; s1 += v; s2 += v * v;
        }
        sred[0][r][oct] = s1; sred[1][r][oct] = s2;
        __syncthreads();
        if (oct == 0) {
            float t1 = 0.f, t2 = 0.f;
#pragma unroll
            for (int q = 0; q < 8; q++) { t1 += sred[0][r][q]; t2 += sred[1][r][q]; }
            float mu = t1 * (1.0f / NHID);
            smean[r] = mu;
            srstd[r] = rsqrtf(t2 * (1.0f / NHID) - mu * mu + EPS_);
        }
    }
    __syncthreads();

    // ---- 128 -> 10 head + mean over forests (atomic accumulate) ----
    if (tid < 32 * NCLASS) {
        int c  = tid % NCLASS, rr = tid / NCLASS;
        float mu = smean[rr], rs = srstd[rr];
        float acc = lin2_b[c];
#pragma unroll 4
        for (int k = 0; k < NHID; k++) {
            float z = (sF[rr][k] - mu) * rs * sg2[k] + sb2g[k];
            acc += z * lin2_w[k * NCLASS + c];
        }
        atomicAdd(&out[(b0 + rr) * NCLASS + c], acc * (1.0f / NFOREST));
    }
}

// ---------------------------------------------------------------------------
extern "C" void kernel_launch(void* const* d_in, const int* in_sizes, int n_in,
                              void* d_out, int out_size, void* d_ws, size_t ws_size,
                              hipStream_t stream) {
    const float* x      = (const float*)d_in[0];
    const float* w1     = (const float*)d_in[1];
    const float* b1     = (const float*)d_in[2];
    const int*   perm   = (const int*)  d_in[3];
    const float* gn1_g  = (const float*)d_in[4];
    const float* gn1_b  = (const float*)d_in[5];
    const float* w2a    = (const float*)d_in[6];
    const float* b2a    = (const float*)d_in[7];
    const float* gn2_g  = (const float*)d_in[8];
    const float* gn2_b  = (const float*)d_in[9];
    const float* w2b    = (const float*)d_in[10];
    const float* b2b    = (const float*)d_in[11];
    const float* E      = (const float*)d_in[12];
    const int*   swr    = (const int*)  d_in[13];
    const float* ln1_g  = (const float*)d_in[14];
    const float* ln1_b  = (const float*)d_in[15];
    const float* lin1_w = (const float*)d_in[16];
    const float* lin1_b = (const float*)d_in[17];
    const float* ln2_g  = (const float*)d_in[18];
    const float* ln2_b  = (const float*)d_in[19];
    const float* lin2_w = (const float*)d_in[20];
    const float* lin2_b = (const float*)d_in[21];
    float* out = (float*)d_out;

    // workspace carve-up (256B aligned)
    char* ws = (char*)d_ws;
    auto align256 = [](size_t v) { return (v + 255) & ~(size_t)255; };
    float* wp   = (float*)ws;                 ws += align256(NCOL * NCOND * 4);
    float* bp   = (float*)ws;                 ws += align256(NCOL * NCOND * 4);
    int*   cix  = (int*)ws;                   ws += align256(NCOL * NCOND * 4);
    float* wbuf = (float*)ws;                 ws += align256((size_t)B_ * NRODT * 4);
    _Float16* EsT = (_Float16*)ws;            ws += align256((size_t)NFOREST * NHID * NEST * 2);
    _Float16* W1T = (_Float16*)ws;            ws += align256((size_t)NHID * NHID * 2);

    k_prep_perm<<<(NCOL * NCOND + 255) / 256, 256, 0, stream>>>(perm, w1, b1, wp, bp, cix);
    k_prep_est<<<(NFOREST * NHID * NEST + 255) / 256, 256, 0, stream>>>(E, swr, EsT);
    k_prep_w1t<<<(NHID * NHID + 255) / 256, 256, 0, stream>>>(lin1_w, W1T);

    k_rodt<<<(B_ * NRODT + 255) / 256, 256, 0, stream>>>(
        x, wp, bp, cix, gn1_g, gn1_b, w2a, b2a, gn2_g, gn2_b, w2b, b2b, wbuf);

    k_zero<<<(B_ * NCLASS + 255) / 256, 256, 0, stream>>>(out, B_ * NCLASS);

    dim3 grid(NFOREST, B_ / 32);
    k_forest<<<grid, 256, 0, stream>>>(wbuf, swr, EsT, W1T,
                                       ln1_g, ln1_b, lin1_b,
                                       ln2_g, ln2_b, lin2_w, lin2_b, out);
}
